// LIF_16484084483619
// MI455X (gfx1250) — compile-verified, matrix operand order checked by reading, not run on hardware
//
#include <hip/hip_runtime.h>
#include <math.h>

// ---------------- problem geometry ----------------
#define TSTEPS 5
#define BDIM   32
#define CDIM   128
#define HWDIM  1024                       // 32*32 contiguous per (b,c) plane
#define NPT    (BDIM * CDIM * HWDIM)      // 4,194,304 elements per timestep
#define PLANES (BDIM * CDIM)              // 4096 blocks, one per (b,c)

#define V_TH    0.5f
#define W_DECAY 0.5f
#define LOWER_C 0.17f                     // 0.2 - 0.03
#define UPPER_C 0.23f                     // 0.2 + 0.03
#define EMA0    0.17f

typedef __attribute__((ext_vector_type(16))) _Float16 v16h;
typedef __attribute__((ext_vector_type(8)))  float    v8f;
typedef __attribute__((ext_vector_type(4)))  float    vf4;   // native vector: OK for nontemporal builtins

__device__ __forceinline__ float sigmoidf_(float x) {
    return 1.0f / (1.0f + expf(-x));
}

// -------- per-timestep element-wise LIF step + WMMA spike-count reduction ----
__global__ void __launch_bounds__(256)
lif_step_kernel(const float* __restrict__ xe_t,
                const float* __restrict__ xi_t,
                const float* __restrict__ alpha_raw,
                const float* __restrict__ beta_raw,
                const float* __restrict__ inhw_c,
                float* __restrict__ mem,
                float* __restrict__ out_t,
                int*   __restrict__ spike_cnt,
                int first)
{
    const int  plane = blockIdx.x;            // plane = b*C + c
    const int  c     = plane & (CDIM - 1);
    const long base  = (long)plane * HWDIM + (long)threadIdx.x * 4;

    const float alpha = 4.0f * sigmoidf_(alpha_raw[0]);
    const float beta  = sigmoidf_(beta_raw[0]);
    const float binh  = beta * (1.0f - inhw_c[c]);   // beta*(1 - inh_weight[c])

    // Streaming inputs: read exactly once -> non-temporal so they do not
    // evict the carried `mem` state from the 192 MB L2.
    const vf4 xe = __builtin_nontemporal_load((const vf4*)(xe_t + base));
    const vf4 xi = __builtin_nontemporal_load((const vf4*)(xi_t + base));
    vf4 m;
    if (first) m = (vf4){0.0f, 0.0f, 0.0f, 0.0f};
    else       m = *(const vf4*)(mem + base);         // regular: L2-resident state

    vf4 sp;
#define LIF_COMP(f)                                              \
    {                                                            \
        float e  = xe.f / (1.0f + alpha * xi.f);                 \
        float mm = W_DECAY * m.f + e - binh * xi.f;              \
        float s  = (mm - V_TH >= 0.0f) ? 1.0f : 0.0f;            \
        sp.f = s;                                                \
        m.f  = mm - V_TH * s;                                    \
    }
    LIF_COMP(x) LIF_COMP(y) LIF_COMP(z) LIF_COMP(w)
#undef LIF_COMP

    // Spike output is write-once-never-read: non-temporal store.
    __builtin_nontemporal_store(sp, (vf4*)(out_t + base));
    *(vf4*)(mem + base) = m;                          // keep state in L2

    // ---- wave spike count via v_wmma_f32_16x16x32_f16 ----
    // A holds each lane's 4 spikes (exact in f16), B = all-ones, C = 0.
    // D[m][n] = row-sum of A, identical in every column n; lanes n and n^16
    // together cover all 16 rows, so s8 + shfl_xor(s8,16) = exact wave total
    // (uniform across the wave).
    v16h a = {};
    a[0] = (_Float16)sp.x; a[1] = (_Float16)sp.y;
    a[2] = (_Float16)sp.z; a[3] = (_Float16)sp.w;
    v16h ones;
#pragma unroll
    for (int i = 0; i < 16; ++i) ones[i] = (_Float16)1.0f;
    v8f acc = {};
    acc = __builtin_amdgcn_wmma_f32_16x16x32_f16(
        /*neg_a=*/false, a, /*neg_b=*/false, ones,
        /*c_mod=*/(short)0, acc, /*reuse_a=*/false, /*reuse_b=*/false);

    float s8 = acc[0] + acc[1] + acc[2] + acc[3] +
               acc[4] + acc[5] + acc[6] + acc[7];
    float wave_tot = s8 + __shfl_xor(s8, 16, 32);     // uniform per wave

    // Block reduction: plain ds_store per wave leader (wave_tot is uniform),
    // thread 0 sums 8 slots, one integer global atomic per block (exact,
    // order-independent -> deterministic).
    __shared__ float wsum[8];
    const int wid = threadIdx.x >> 5;
    if ((threadIdx.x & 31) == 0) wsum[wid] = wave_tot;
    __syncthreads();
    if (threadIdx.x == 0) {
        float t = 0.0f;
#pragma unroll
        for (int i = 0; i < 8; ++i) t += wsum[i];
        atomicAdd(&spike_cnt[c], (int)(t + 0.5f));
    }
}

// -------- per-channel EMA / inhibition-weight update between timesteps ------
__global__ void lif_channel_kernel(float* __restrict__ ema,
                                   float* __restrict__ inhw,
                                   int*   __restrict__ spike_cnt)
{
    const int c = threadIdx.x;                      // 128 threads
    const float mean = (float)spike_cnt[c] * (1.0f / (float)(BDIM * HWDIM));
    const float e    = 0.9f * ema[c] + 0.1f * mean;
    ema[c]  = e;
    inhw[c] = 4.0f * (sigmoidf_(LOWER_C - e) - sigmoidf_(e - UPPER_C));
    spike_cnt[c] = 0;                               // ready for next timestep
}

__global__ void lif_init_kernel(float* __restrict__ ema,
                                float* __restrict__ inhw,
                                int*   __restrict__ spike_cnt)
{
    const int c = threadIdx.x;
    ema[c]  = EMA0;
    inhw[c] = 0.0f;
    spike_cnt[c] = 0;
}

extern "C" void kernel_launch(void* const* d_in, const int* in_sizes, int n_in,
                              void* d_out, int out_size, void* d_ws, size_t ws_size,
                              hipStream_t stream)
{
    const float* xe        = (const float*)d_in[0];   // [T,B,C,H,W]
    const float* xi        = (const float*)d_in[1];   // [T,B,C,H,W]
    const float* alpha_raw = (const float*)d_in[2];   // scalar
    const float* beta_raw  = (const float*)d_in[3];   // scalar
    float*       out       = (float*)d_out;           // [T,B,C,H,W]

    // workspace layout: mem state (16 MB) | ema[128] | inhw[128] | cnt[128]
    char*  ws   = (char*)d_ws;
    float* mem  = (float*)ws;
    float* ema  = (float*)(ws + (size_t)NPT * sizeof(float));
    float* inhw = ema + CDIM;
    int*   cnt  = (int*)(inhw + CDIM);

    lif_init_kernel<<<1, CDIM, 0, stream>>>(ema, inhw, cnt);

    for (int t = 0; t < TSTEPS; ++t) {
        lif_step_kernel<<<PLANES, 256, 0, stream>>>(
            xe + (size_t)t * NPT, xi + (size_t)t * NPT,
            alpha_raw, beta_raw, inhw, mem,
            out + (size_t)t * NPT, cnt, (t == 0) ? 1 : 0);
        lif_channel_kernel<<<1, CDIM, 0, stream>>>(ema, inhw, cnt);
    }
}
